// ContextAttention_68959994904720
// MI455X (gfx1250) — matrix operand order repaired and checked by reading, not compiled, not run on hardware
//
#include <hip/hip_runtime.h>
#include <stdint.h>

#define NB 8192
#define SS 20
#define DD 300
#define JJ 512
#define NIg 1000
#define VGc 2078
#define KC 320    // D padded to mult of 32
#define KV 2080   // VG padded to mult of 32
#define MG 1008   // NI padded to mult of 16
#define BPB 4     // batch elements per block (fused kernel)
#define MR (BPB * SS)   // 80 rows = 5 exact 16-row tiles

typedef __bf16 bf16_t;
typedef bf16_t v16bf __attribute__((ext_vector_type(16)));
typedef float  v8f   __attribute__((ext_vector_type(8)));
typedef unsigned int v4u __attribute__((ext_vector_type(4)));

union FragU { v16bf v; v4u q[2]; };

__device__ __forceinline__ unsigned short f32_bf16(float f) {
  unsigned int x = __float_as_uint(f);
  unsigned int r = (x + 0x7FFFu + ((x >> 16) & 1u)) >> 16;  // RNE
  return (unsigned short)r;
}

// ---- prep: pad+convert f32 matrix -> bf16 (zero padded) ----
__global__ void pad_bf16(const float* __restrict__ src, unsigned short* __restrict__ dst,
                         int rs, int cs, int rd, int cd) {
  int stride = gridDim.x * blockDim.x;
  int total = rd * cd;
  for (int idx = blockIdx.x * blockDim.x + threadIdx.x; idx < total; idx += stride) {
    int r = idx / cd, c = idx - r * cd;
    float v = (r < rs && c < cs) ? src[(size_t)r * cs + c] : 0.f;
    dst[idx] = f32_bf16(v);
  }
}

__global__ void make_bsum(const float* __restrict__ bs, const float* __restrict__ be,
                          float* __restrict__ bsum) {
  int j = blockIdx.x * blockDim.x + threadIdx.x;
  if (j < JJ) bsum[j] = bs[j] + be[j];
}

// ---- P[1000][512] = guide(bf16,padded) @ Wemb(bf16,padded)^T ----
__global__ __launch_bounds__(256) void proj_gemm(const unsigned short* __restrict__ Gb,
                                                 const unsigned short* __restrict__ Wb,
                                                 float* __restrict__ P) {
  const int tid = threadIdx.x;
  const int wave = tid >> 5, lane = tid & 31;
  const int lh = lane & 15, hi = lane >> 4;
  const int r0 = blockIdx.x * 16;      // 63 blocks cover 1008 rows
  const int jb = wave * 64;            // 8 waves * 64 cols = 512
  v8f acc[4] = {};
  for (int kk = 0; kk < KV / 32; ++kk) {
    FragU a;
    const unsigned short* ap = Gb + (size_t)(r0 + lh) * KV + kk * 32 + hi * 8;
    a.q[0] = *(const v4u*)ap;          // A layout: K = koff+0..7
    a.q[1] = *(const v4u*)(ap + 16);   //           K = koff+16..23
#pragma unroll
    for (int nt = 0; nt < 4; ++nt) {
      FragU bb;
      const unsigned short* bp = Wb + (size_t)(jb + nt * 16 + lh) * KV + kk * 32 + hi * 8;
      bb.q[0] = *(const v4u*)bp;       // B layout: K = koff+0..7, koff+16..23 (same as A)
      bb.q[1] = *(const v4u*)(bp + 16);
      acc[nt] = __builtin_amdgcn_wmma_f32_16x16x32_bf16(false, a.v, false, bb.v,
                                                        (short)0, acc[nt], false, false);
    }
  }
#pragma unroll
  for (int nt = 0; nt < 4; ++nt)
#pragma unroll
    for (int v = 0; v < 8; ++v) {
      int r = r0 + v + 8 * hi;         // C/D layout: row = vgpr + 8*(lane>=16)
      int j = jb + nt * 16 + lh;
      if (r < NIg) P[(size_t)r * JJ + j] = acc[nt][v];
    }
}

// ---- fused: 4 batch elems/block, 80x512 WMMA GEMM + tanh + masked softmax + weighted sum ----
__global__ __launch_bounds__(512) void fused_attn(
    const float* __restrict__ ctx, const float* __restrict__ emb,
    const int* __restrict__ labels, const int* __restrict__ ann,
    const unsigned short* __restrict__ Wsb, const float* __restrict__ P,
    const float* __restrict__ bsum, const float* __restrict__ wfc,
    const float* __restrict__ bfc, float* __restrict__ out) {
  __shared__ __align__(16) unsigned short sA[MR * KC];   // 50 KB: 80 ctx rows, bf16
  __shared__ float sProw[BPB][JJ];                       // 8 KB
  __shared__ float sBsum[JJ], sWfc[JJ];
  __shared__ float sScore[BPB * 32];
  __shared__ float sAttn[BPB][SS], sValid[BPB][SS], sNon[BPB][SS];

  const int b0 = blockIdx.x * BPB;
  const int tid = threadIdx.x;
  const int wave = tid >> 5, lane = tid & 31;
  const int lh = lane & 15, hi = lane >> 4;

  // stage 4 context blocks as bf16 [80][320]; row r -> (g = r/20, s = r%20)
  for (int idx = tid; idx < MR * KC; idx += 512) {
    int r = idx / KC, k = idx - r * KC;
    float v = (k < DD) ? ctx[(size_t)(b0 + r / SS) * (SS * DD) + (r % SS) * DD + k] : 0.f;
    sA[idx] = f32_bf16(v);
  }
  for (int j = tid; j < JJ; j += 512) {
    sBsum[j] = bsum[j];
    sWfc[j]  = wfc[j];
  }
  for (int idx = tid; idx < BPB * JJ; idx += 512) {
    int g = idx >> 9, j = idx & (JJ - 1);
    sProw[g][j] = P[(size_t)ann[2 * (b0 + g)] * JJ + j];
  }
  if (tid < BPB * 32) sScore[tid] = 0.f;
  if (tid < BPB) {
    int g = tid;
    float v = 1.f;
    for (int s = 0; s < SS; ++s) {
      float nz = (labels[(b0 + g) * SS + s] != 0) ? 1.f : 0.f;
      v *= nz;
      sNon[g][s] = nz; sValid[g][s] = v;
    }
  }
  __syncthreads();

  // GEMM: each of 16 waves owns 32 columns (2 N-tiles) x 5 M-tiles
  const int jb = wave * 32;
  v8f acc[5][2] = {};
  for (int kk = 0; kk < KC / 32; ++kk) {
    FragU a[5];
#pragma unroll
    for (int mt = 0; mt < 5; ++mt) {
      const unsigned short* ap = &sA[(mt * 16 + lh) * KC + kk * 32 + hi * 8];
      a[mt].q[0] = *(const v4u*)ap;
      a[mt].q[1] = *(const v4u*)(ap + 16);
    }
#pragma unroll
    for (int nt = 0; nt < 2; ++nt) {
      FragU bb;
      const unsigned short* bp = Wsb + (size_t)(jb + nt * 16 + lh) * KC + kk * 32 + hi * 8;
      bb.q[0] = *(const v4u*)bp;
      bb.q[1] = *(const v4u*)(bp + 16);
#pragma unroll
      for (int mt = 0; mt < 5; ++mt)
        acc[mt][nt] = __builtin_amdgcn_wmma_f32_16x16x32_bf16(false, a[mt].v, false, bb.v,
                                                              (short)0, acc[mt][nt], false, false);
    }
  }

  // epilogue: H = tanh(acc + bsum + valid*P) ; per-(g,s) score partials
#pragma unroll
  for (int mt = 0; mt < 5; ++mt) {
#pragma unroll
    for (int v = 0; v < 8; ++v) {
      int r = mt * 16 + v + 8 * hi;          // 0..79, all rows real
      int g = r / SS, s = r - g * SS;
      float vs = sValid[g][s];
      float partial = 0.f;
#pragma unroll
      for (int nt = 0; nt < 2; ++nt) {
        int j = jb + nt * 16 + lh;
        float h = tanhf(acc[mt][nt][v] + sBsum[j] + vs * sProw[g][j]);
        partial += h * sWfc[j];
      }
      for (int off = 1; off < 16; off <<= 1)  // reduce across half-wave (same r)
        partial += __shfl_xor(partial, off, 16);
      if (lh == 0) atomicAdd(&sScore[g * 32 + s], partial);
    }
  }
  __syncthreads();

  // masked softmax: waves 0..3, one per batch element
  if (tid < BPB * 32) {
    int g = tid >> 5, s = tid & 31;
    float sc = (s < SS) ? sScore[g * 32 + s] + bfc[0] : -3.0e38f;
    float m = sc;
    for (int off = 1; off < 32; off <<= 1) m = fmaxf(m, __shfl_xor(m, off, 32));
    float e = (s < SS) ? __expf(sc - m) * sNon[g][s] : 0.f;
    float sum = e;
    for (int off = 1; off < 32; off <<= 1) sum += __shfl_xor(sum, off, 32);
    if (s < SS) sAttn[g][s] = e / sum;
  }
  __syncthreads();

  // weighted sum of embedded -> out
  for (int idx = tid; idx < BPB * DD; idx += 512) {
    int g = idx / DD, d = idx - g * DD;
    const float* ep = emb + (size_t)(b0 + g) * (SS * DD) + d;
    float a = 0.f;
#pragma unroll
    for (int s = 0; s < SS; ++s) a += sAttn[g][s] * ep[s * DD];
    out[(size_t)(b0 + g) * DD + d] = a;
  }
}

extern "C" void kernel_launch(void* const* d_in, const int* in_sizes, int n_in,
                              void* d_out, int out_size, void* d_ws, size_t ws_size,
                              hipStream_t stream) {
  const float* ctx    = (const float*)d_in[0];
  const float* emb    = (const float*)d_in[1];
  const int*   labels = (const int*)d_in[2];
  const float* guide  = (const float*)d_in[3];
  const int*   ann    = (const int*)d_in[4];
  const float* Wsent  = (const float*)d_in[5];
  const float* b_sent = (const float*)d_in[6];
  const float* Wemb   = (const float*)d_in[7];
  const float* b_emb  = (const float*)d_in[8];
  const float* wfc    = (const float*)d_in[9];
  const float* bfc    = (const float*)d_in[10];
  float* out = (float*)d_out;

  char* ws = (char*)d_ws;
  size_t off = 0;
  auto alloc = [&](size_t bytes) -> char* {
    char* p = ws + off;
    off = (off + bytes + 255) & ~(size_t)255;
    return p;
  };
  unsigned short* Wsb  = (unsigned short*)alloc((size_t)JJ * KC * 2);   // 320 KB
  unsigned short* Webb = (unsigned short*)alloc((size_t)JJ * KV * 2);   // 2.1 MB
  unsigned short* Gb   = (unsigned short*)alloc((size_t)MG * KV * 2);   // 4.2 MB
  float* P    = (float*)alloc((size_t)NIg * JJ * 4);                    // 2.0 MB
  float* bsum = (float*)alloc((size_t)JJ * 4);

  pad_bf16<<<640, 256, 0, stream>>>(Wsent, Wsb, JJ, DD, JJ, KC);
  pad_bf16<<<2048, 256, 0, stream>>>(Wemb, Webb, JJ, VGc, JJ, KV);
  pad_bf16<<<2048, 256, 0, stream>>>(guide, Gb, NIg, VGc, MG, KV);
  make_bsum<<<2, 256, 0, stream>>>(b_sent, b_emb, bsum);
  proj_gemm<<<MG / 16, 256, 0, stream>>>(Gb, Webb, P);
  fused_attn<<<NB / BPB, 512, 0, stream>>>(ctx, emb, labels, ann, Wsb, P, bsum, wfc, bfc, out);
}
